// BusNode_8581344657619
// MI455X (gfx1250) — matrix operand order repaired and checked by reading, not compiled, not run on hardware
//
#include <hip/hip_runtime.h>
#include <hip/hip_bf16.h>

typedef __attribute__((ext_vector_type(16))) __bf16 v16bf;
typedef __attribute__((ext_vector_type(8)))  __bf16 v8bf;
typedef __attribute__((ext_vector_type(8)))  float  v8f;
typedef __attribute__((ext_vector_type(4)))  float  v4f;

#define T_   8
#define N_   8192        // B*S
#define LAT  1024
#define SYM  256
#define NC   512

// ---------------------------------------------------------------------------
// 1) Weight pack: N-major bf16 weights (coalesced B staging), bf16 codebook,
//    fp32 codebook^T + ||c||^2.
// ---------------------------------------------------------------------------
__global__ __launch_bounds__(256) void pack_kernel(
    const float* __restrict__ Wread, const float* __restrict__ Wsym,
    const float* __restrict__ Wc1,   const float* __restrict__ Wc2,
    const float* __restrict__ codebook,
    __bf16* __restrict__ Wt1, __bf16* __restrict__ Wt2,
    __bf16* __restrict__ Wt3, __bf16* __restrict__ Wt4,
    __bf16* __restrict__ cb_bf, float* __restrict__ cbT,
    float* __restrict__ cnorm)
{
  long i = (long)blockIdx.x * 256 + threadIdx.x;
  const long n1 = (long)LAT * (2 * LAT);
  if (i < n1) { long n = i / (2*LAT), k = i % (2*LAT); Wt1[i] = (__bf16)Wread[k * LAT + n]; return; }
  i -= n1;
  const long n2 = (long)SYM * LAT;
  if (i < n2) { long n = i / LAT, k = i % LAT; Wt2[i] = (__bf16)Wsym[k * SYM + n]; return; }
  i -= n2;
  const long n3 = (long)LAT * (LAT + SYM);
  if (i < n3) { long n = i / (LAT+SYM), k = i % (LAT+SYM); Wt3[i] = (__bf16)Wc1[k * LAT + n]; return; }
  i -= n3;
  const long n4 = (long)LAT * LAT;
  if (i < n4) { long n = i / LAT, k = i % LAT; Wt4[i] = (__bf16)Wc2[k * LAT + n]; return; }
  i -= n4;
  const long n5 = (long)NC * SYM;
  if (i < n5) { cb_bf[i] = (__bf16)codebook[i]; return; }
  i -= n5;
  const long n6 = (long)SYM * NC;                 // cbT[k*NC + c] = codebook[c*SYM + k]
  if (i < n6) { long k = i / NC, c = i % NC; cbT[i] = codebook[c * SYM + k]; return; }
  i -= n6;
  if (i < NC) {
    float s = 0.f;
    for (int k = 0; k < SYM; ++k) { float v = codebook[i * SYM + k]; s = fmaf(v, v, s); }
    cnorm[i] = s;
  }
}

// ---------------------------------------------------------------------------
// 2) Routing: one wave32 per token; butterfly-reduced dot, first-max argmax.
// ---------------------------------------------------------------------------
__global__ __launch_bounds__(256) void route_kernel(
    const float* __restrict__ syms, const float* __restrict__ Wq,
    int* __restrict__ top)
{
  const int wid = threadIdx.x >> 5, lane = threadIdx.x & 31;
  const int n = blockIdx.x * 8 + wid;
  float wq[8];
#pragma unroll
  for (int j = 0; j < 8; ++j) wq[j] = Wq[lane + 32 * j];
  float best = -3.4e38f; int bt = 0;
#pragma unroll
  for (int t = 0; t < T_; ++t) {
    const float* p = syms + ((long)t * N_ + n) * SYM;
    float acc = 0.f;
#pragma unroll
    for (int j = 0; j < 8; ++j) acc = fmaf(p[lane + 32 * j], wq[j], acc);
#pragma unroll
    for (int s = 16; s > 0; s >>= 1) acc += __shfl_xor(acc, s, 32);
    if (acc > best) { best = acc; bt = t; }
  }
  if (lane == 0) top[n] = bt;
}

// ---------------------------------------------------------------------------
// 3) WMMA GEMM, block tile 128x256x32, 8 waves (2M x 4N), wave tile 64x64 =
//    4x4 fragments of v_wmma_f32_16x16x32_bf16. Branch-free inner loop:
//    gathers are resolved ONCE per thread into row-base pointers and the
//    K-range is split into homogeneous segments.
// ---------------------------------------------------------------------------
constexpr int BM = 128, BN = 256, BK = 32;
constexpr int ASTR = 40;   // padded bf16 row stride -> conflict-free ds_load_b128
constexpr int BSTR = 40;

// load 8 contiguous elements as bf16 (converting from fp32 if needed)
template <typename T>
__device__ __forceinline__ v8bf load8_bf(const T* p) {
  if constexpr (sizeof(T) == 4) {
    v4f x0 = *(const v4f*)p;
    v4f x1 = *(const v4f*)(p + 4);
    v8bf o;
#pragma unroll
    for (int j = 0; j < 4; ++j) { o[j] = (__bf16)x0[j]; o[4 + j] = (__bf16)x1[j]; }
    return o;
  } else {
    return *(const v8bf*)p;
  }
}

template <int MODE>
__global__ __launch_bounds__(256) void gemm_wmma(
    const float*  __restrict__ A0f, const __bf16* __restrict__ A0b,
    const float*  __restrict__ A1f, const __bf16* __restrict__ A1b,
    const int*    __restrict__ gidx,
    const __bf16* __restrict__ Wt,  const float* __restrict__ bias,
    float* __restrict__ Cf, __bf16* __restrict__ Cb,
    const float* __restrict__ resid,
    int Ndim, int Ktot)
{
  __shared__ __bf16 As[BM * ASTR];
  __shared__ __bf16 Bs[BN * BSTR];

  const int tid = threadIdx.x;
  const int m0  = blockIdx.y * BM;
  const int n0  = blockIdx.x * BN;
  const int wid = tid >> 5, lane = tid & 31;
  const int wm  = (wid >> 2) * 64;     // 0,64
  const int wn  = (wid & 3) * 64;      // 0,64,128,192
  const int lg  = lane >> 4;           // lane group 0/1
  const int lr  = lane & 15;

  // --- fixed per-thread staging geometry ---
  const int ar0 = tid >> 2, ar1 = (tid >> 2) + 64;     // A rows (2 chunks)
  const int ak8 = (tid & 3) * 8;                       // A k-offset within tile
  __bf16* asd0 = &As[ar0 * ASTR + ak8];
  __bf16* asd1 = &As[ar1 * ASTR + ak8];
  const int bk8 = (tid & 3) * 8;
  const __bf16* wb[4];
  __bf16* bsd[4];
#pragma unroll
  for (int i = 0; i < 4; ++i) {
    const int nn = (tid >> 2) + 64 * i;
    wb[i]  = Wt + (long)(n0 + nn) * Ktot + bk8;
    bsd[i] = &Bs[nn * BSTR + bk8];
  }

  v8f acc[4][4];
#pragma unroll
  for (int i = 0; i < 4; ++i)
#pragma unroll
    for (int j = 0; j < 4; ++j)
      acc[i][j] = v8f{0.f, 0.f, 0.f, 0.f, 0.f, 0.f, 0.f, 0.f};

  // homogeneous K-segment: rb0/rb1 are per-thread A row-base pointers with
  // ak8 folded in; rb + kb is the load address (kb absolute).
  auto seg = [&](const auto* rb0, const auto* rb1, int k0, int k1) {
    for (int kb = k0; kb < k1; kb += BK) {
      *(v8bf*)asd0 = load8_bf(rb0 + kb);
      *(v8bf*)asd1 = load8_bf(rb1 + kb);
#pragma unroll
      for (int i = 0; i < 4; ++i)
        *(v8bf*)bsd[i] = *(const v8bf*)(wb[i] + kb);
      if (kb + BK < Ktot) __builtin_prefetch(wb[0] + kb + BK, 0, 1);
      __syncthreads();

      // A frags: lane row = lr, K chunks [8*lg,+8) and [8*lg+16,+8)
      v16bf af[4];
#pragma unroll
      for (int mf = 0; mf < 4; ++mf) {
        const __bf16* pa = &As[(wm + mf * 16 + lr) * ASTR + lg * 8];
        v8bf lo = *(const v8bf*)pa;
        v8bf hi = *(const v8bf*)(pa + 16);
        af[mf] = __builtin_shufflevector(lo, hi, 0,1,2,3,4,5,6,7,8,9,10,11,12,13,14,15);
      }
      // B frags one at a time: lane col = lr, K chunks [8*lg,+8) and [8*lg+16,+8)
#pragma unroll
      for (int nf = 0; nf < 4; ++nf) {
        const __bf16* pb = &Bs[(wn + nf * 16 + lr) * BSTR + lg * 8];
        v8bf lo = *(const v8bf*)pb;
        v8bf hi = *(const v8bf*)(pb + 16);
        v16bf bfr = __builtin_shufflevector(lo, hi, 0,1,2,3,4,5,6,7,8,9,10,11,12,13,14,15);
#pragma unroll
        for (int mf = 0; mf < 4; ++mf)
          acc[mf][nf] = __builtin_amdgcn_wmma_f32_16x16x32_bf16(
              false, af[mf], false, bfr, (short)0, acc[mf][nf], false, false);
      }
      __syncthreads();
    }
  };

  if constexpr (MODE == 0) {
    // seg1: token_state fp32 (stride LAT); seg2: gathered bus_outputs fp32
    const float* rb0 = A0f + (long)(m0 + ar0) * LAT + ak8;
    const float* rb1 = A0f + (long)(m0 + ar1) * LAT + ak8;
    seg(rb0, rb1, 0, LAT);
    const float* gb0 = A1f + ((long)gidx[m0 + ar0] * N_ + (m0 + ar0)) * LAT + ak8 - LAT;
    const float* gb1 = A1f + ((long)gidx[m0 + ar1] * N_ + (m0 + ar1)) * LAT + ak8 - LAT;
    seg(gb0, gb1, LAT, 2 * LAT);
  } else if constexpr (MODE == 2) {
    // seg1: z_read bf16 (stride LAT); seg2: gathered codebook bf16
    const __bf16* rb0 = A0b + (long)(m0 + ar0) * LAT + ak8;
    const __bf16* rb1 = A0b + (long)(m0 + ar1) * LAT + ak8;
    seg(rb0, rb1, 0, LAT);
    const __bf16* gb0 = A1b + (long)gidx[m0 + ar0] * SYM + ak8 - LAT;
    const __bf16* gb1 = A1b + (long)gidx[m0 + ar1] * SYM + ak8 - LAT;
    seg(gb0, gb1, LAT, LAT + SYM);
  } else {
    const __bf16* rb0 = A0b + (long)(m0 + ar0) * Ktot + ak8;
    const __bf16* rb1 = A0b + (long)(m0 + ar1) * Ktot + ak8;
    seg(rb0, rb1, 0, Ktot);
  }

  // --- epilogue: C layout: VGPR r -> row = 8*lg + r, lane lr -> col ---
#pragma unroll
  for (int nf = 0; nf < 4; ++nf) {
    const int col = n0 + wn + nf * 16 + lr;
    const float bv = bias[col];
#pragma unroll
    for (int mf = 0; mf < 4; ++mf) {
#pragma unroll
      for (int r = 0; r < 8; ++r) {
        const int row = m0 + wm + mf * 16 + lg * 8 + r;
        const long o = (long)row * Ndim + col;
        const float v = acc[mf][nf][r] + bv;
        if constexpr (MODE == 0)       Cb[o] = (__bf16)v;
        else if constexpr (MODE == 1)  Cf[o] = v;
        else if constexpr (MODE == 2)  Cb[o] = (__bf16)fmaxf(v, 0.f);
        else                           Cf[o] = v + resid[o];
      }
    }
  }
}

// ---------------------------------------------------------------------------
// 4) VQ: one 512-thread block per token; argmin_c ||c||^2 - 2 f.c,
//    first-index tie-break; writes quantized + index straight to d_out.
// ---------------------------------------------------------------------------
__global__ __launch_bounds__(512) void vq_kernel(
    const float* __restrict__ rawsym, const float* __restrict__ cbT,
    const float* __restrict__ cnorm,  const float* __restrict__ codebook,
    int* __restrict__ idx_out, float* __restrict__ q_out,
    float* __restrict__ idxf_out)
{
  __shared__ float rs[SYM];
  __shared__ float dmin[512];
  __shared__ int   didx[512];
  const int n = blockIdx.x, t = threadIdx.x;
  if (t < SYM) rs[t] = rawsym[(long)n * SYM + t];
  __syncthreads();
  float dot = 0.f;
#pragma unroll 8
  for (int k = 0; k < SYM; ++k) dot = fmaf(rs[k], cbT[(long)k * NC + t], dot);
  dmin[t] = cnorm[t] - 2.f * dot;
  didx[t] = t;
  __syncthreads();
  for (int s = 256; s > 0; s >>= 1) {
    if (t < s) {
      const float a = dmin[t], b = dmin[t + s];
      if (b < a || (b == a && didx[t + s] < didx[t])) { dmin[t] = b; didx[t] = didx[t + s]; }
    }
    __syncthreads();
  }
  const int best = didx[0];
  if (t == 0) { idx_out[n] = best; idxf_out[n] = (float)best; }
  if (t < SYM) q_out[(long)n * SYM + t] = codebook[(long)best * SYM + t];
}

// ---------------------------------------------------------------------------
// 5) keep_mask
// ---------------------------------------------------------------------------
__global__ __launch_bounds__(256) void finalize_kernel(
    const int* __restrict__ top, float* __restrict__ keep)
{
  __shared__ unsigned m[256];
  unsigned b = 0;
  for (int n = threadIdx.x; n < N_; n += 256) b |= 1u << top[n];
  m[threadIdx.x] = b;
  __syncthreads();
  for (int s = 128; s > 0; s >>= 1) {
    if (threadIdx.x < s) m[threadIdx.x] |= m[threadIdx.x + s];
    __syncthreads();
  }
  if (threadIdx.x < T_) keep[threadIdx.x] = ((m[0] >> threadIdx.x) & 1u) ? 0.f : 1.f;
}

// ---------------------------------------------------------------------------
extern "C" void kernel_launch(void* const* d_in, const int* in_sizes, int n_in,
                              void* d_out, int out_size, void* d_ws, size_t ws_size,
                              hipStream_t stream) {
  const float* token_state = (const float*)d_in[0];
  const float* bus_symbols = (const float*)d_in[1];
  const float* bus_outputs = (const float*)d_in[3];
  const float* Wq    = (const float*)d_in[5];
  const float* Wread = (const float*)d_in[7];
  const float* bread = (const float*)d_in[8];
  const float* Wsym  = (const float*)d_in[9];
  const float* bsym  = (const float*)d_in[10];
  const float* Wc1   = (const float*)d_in[11];
  const float* bc1   = (const float*)d_in[12];
  const float* Wc2   = (const float*)d_in[13];
  const float* bc2   = (const float*)d_in[14];
  const float* codebook = (const float*)d_in[15];

  char* p = (char*)d_ws;
  auto alloc = [&](size_t bytes) -> void* {
    void* r = (void*)p;
    p += (bytes + 255) & ~(size_t)255;
    return r;
  };
  __bf16* Wt1   = (__bf16*)alloc((size_t)LAT * 2 * LAT * 2);
  __bf16* Wt2   = (__bf16*)alloc((size_t)SYM * LAT * 2);
  __bf16* Wt3   = (__bf16*)alloc((size_t)LAT * (LAT + SYM) * 2);
  __bf16* Wt4   = (__bf16*)alloc((size_t)LAT * LAT * 2);
  __bf16* cb_bf = (__bf16*)alloc((size_t)NC * SYM * 2);
  float*  cbT   = (float*)alloc((size_t)SYM * NC * 4);
  float*  cnorm = (float*)alloc((size_t)NC * 4);
  int*    top   = (int*)alloc((size_t)N_ * 4);
  int*    vqidx = (int*)alloc((size_t)N_ * 4);
  __bf16* zread = (__bf16*)alloc((size_t)N_ * LAT * 2);
  float*  rawsm = (float*)alloc((size_t)N_ * SYM * 4);
  __bf16* hbuf  = (__bf16*)alloc((size_t)N_ * LAT * 2);

  float* out_node = (float*)d_out;                 // [N_, LAT]
  float* out_q    = out_node + (long)N_ * LAT;     // [N_, SYM]
  float* out_idx  = out_q + (long)N_ * SYM;        // [N_]
  float* out_keep = out_idx + N_;                  // [T_]

  {
    const long total = (long)LAT*2*LAT + (long)SYM*LAT + (long)LAT*(LAT+SYM)
                     + (long)LAT*LAT + (long)NC*SYM + (long)SYM*NC + NC;
    const int grid = (int)((total + 255) / 256);
    pack_kernel<<<grid, 256, 0, stream>>>(Wread, Wsym, Wc1, Wc2, codebook,
                                          Wt1, Wt2, Wt3, Wt4, cb_bf, cbT, cnorm);
  }
  route_kernel<<<N_ / 8, 256, 0, stream>>>(bus_symbols, Wq, top);

  // z_read = [token_state | gather(bus_outputs, top)] @ Wread + bread -> bf16
  gemm_wmma<0><<<dim3(LAT / BN, N_ / BM), 256, 0, stream>>>(
      token_state, nullptr, bus_outputs, nullptr, top,
      Wt1, bread, nullptr, zread, nullptr, LAT, 2 * LAT);

  // raw_symbol = z_read @ Wsym + bsym -> f32
  gemm_wmma<1><<<dim3(SYM / BN, N_ / BM), 256, 0, stream>>>(
      nullptr, zread, nullptr, nullptr, nullptr,
      Wt2, bsym, rawsm, nullptr, nullptr, SYM, LAT);

  vq_kernel<<<N_, 512, 0, stream>>>(rawsm, cbT, cnorm, codebook, vqidx, out_q, out_idx);

  // h = relu([z_read | gather(codebook)] @ Wc1 + bc1) -> bf16
  gemm_wmma<2><<<dim3(LAT / BN, N_ / BM), 256, 0, stream>>>(
      nullptr, zread, nullptr, cb_bf, vqidx,
      Wt3, bc1, nullptr, hbuf, nullptr, LAT, LAT + SYM);

  // node_output = h @ Wc2 + bc2 + token_state -> d_out
  gemm_wmma<3><<<dim3(LAT / BN, N_ / BM), 256, 0, stream>>>(
      nullptr, hbuf, nullptr, nullptr, nullptr,
      Wt4, bc2, out_node, nullptr, token_state, LAT, LAT);

  finalize_kernel<<<1, 256, 0, stream>>>(top, out_keep);
}
